// MultiHeadLatentAttention_31284541784328
// MI455X (gfx1250) — matrix operand order repaired and checked by reading, not compiled, not run on hardware
//
#include <hip/hip_runtime.h>
#include <hip/hip_bf16.h>

// ---------------------------------------------------------------------------
// CDNA5 async global->LDS path (guarded so the file compiles either way)
// ---------------------------------------------------------------------------
#if defined(__has_builtin)
#  if __has_builtin(__builtin_amdgcn_global_load_async_to_lds_b128)
#    define USE_ASYNC_LDS 1
#  else
#    define USE_ASYNC_LDS 0
#  endif
#else
#  define USE_ASYNC_LDS 0
#endif

// ---------------------------------------------------------------------------
// Types
// ---------------------------------------------------------------------------
typedef unsigned short u16;
typedef __attribute__((ext_vector_type(16))) __bf16   v16bf;
typedef __attribute__((ext_vector_type(8)))  float    v8f;
typedef __attribute__((ext_vector_type(4)))  unsigned v4u;

// Exact pointee type expected by __builtin_amdgcn_global_load_async_to_lds_b128
typedef int v4i __attribute__((vector_size(16)));
typedef __attribute__((address_space(1))) v4i as1_v4i;   // global
typedef __attribute__((address_space(3))) v4i as3_v4i;   // LDS

union Frag {
    v16bf v;
    v4u   q[2];
    u16   u[16];
};

__device__ __forceinline__ u16 f2bf(float f) {
    unsigned u = __float_as_uint(f);
    unsigned r = u + 0x7FFFu + ((u >> 16) & 1u);   // round-to-nearest-even
    return (u16)(r >> 16);
}
__device__ __forceinline__ float bf2f(u16 h) {
    return __uint_as_float(((unsigned)h) << 16);
}

__device__ __forceinline__ v8f wmma_bf16(v16bf a, v16bf b, v8f c) {
    // 8 args: (neg_a, A, neg_b, B, c_mod, C, reuse_a, reuse_b)
    return __builtin_amdgcn_wmma_f32_16x16x32_bf16(false, a, false, b,
                                                   (short)0, c, false, false);
}

// 16-byte global -> LDS copy: async DMA (ASYNCcnt) when available, else
// regular load + ds_store.
__device__ __forceinline__ void copy16_g2l(const u16* g, u16* l) {
#if USE_ASYNC_LDS
    __builtin_amdgcn_global_load_async_to_lds_b128(
        (as1_v4i*)(unsigned long long)g,
        (as3_v4i*)(unsigned)(unsigned long long)l,
        0, 0);
#else
    *reinterpret_cast<v4u*>(l) = *reinterpret_cast<const v4u*>(g);
#endif
}

__device__ __forceinline__ void async_wait_all() {
#if USE_ASYNC_LDS
#  if defined(__has_builtin) && __has_builtin(__builtin_amdgcn_s_wait_asynccnt)
    __builtin_amdgcn_s_wait_asynccnt(0);
#  else
    asm volatile("s_wait_asynccnt 0x0" ::: "memory");
#  endif
#endif
}

// A fragment 16x32 (MxK) from row-major [rows x ld] bf16 (global or LDS).
// Lane l<16: K = kb+[0..7], kb+[16..23]; lane>=16: K = kb+[8..15], kb+[24..31].
__device__ __forceinline__ v16bf load_fragA(const u16* base, int ld, int row0,
                                            int kb, int lane) {
    const u16* p = base + (size_t)(row0 + (lane & 15)) * ld + kb + ((lane >> 4) << 3);
    Frag f;
    f.q[0] = *reinterpret_cast<const v4u*>(p);
    f.q[1] = *reinterpret_cast<const v4u*>(p + 16);
    return f.v;
}

// B fragment 32x16 (KxN) where source is row-major "W" layout [N x K]:
// lane n (0..15) holds column n with K = kb+[0..7], kb+[16..23]; lane n+16: K = kb+[8..15], kb+[24..31].
__device__ __forceinline__ v16bf load_fragB(const u16* base, int ld, int col0,
                                            int kb, int lane) {
    const u16* p = base + (size_t)(col0 + (lane & 15)) * ld + kb + ((lane >> 4) << 3);
    Frag f;
    f.q[0] = *reinterpret_cast<const v4u*>(p);
    f.q[1] = *reinterpret_cast<const v4u*>(p + 16);
    return f.v;
}

// ---------------------------------------------------------------------------
// fp32 -> bf16 convert
// ---------------------------------------------------------------------------
__global__ void cvt_f32_to_bf16_kernel(const float* __restrict__ src,
                                       u16* __restrict__ dst, int n) {
    int i = blockIdx.x * blockDim.x + threadIdx.x;
    if (i < n) dst[i] = f2bf(src[i]);
}

// ---------------------------------------------------------------------------
// LDS-tiled WMMA GEMM: Out[M,N] = A[M,K] * W[N,K]^T  (A, W bf16 row-major)
// Block 256 threads = 8 waves. Block tile 128x128, K-slab 64 staged in LDS
// (async DMA when available). Wave tile 32x64 = 8 WMMA accumulators.
// ---------------------------------------------------------------------------
#define BM  128
#define BN  128
#define BK  64
#define LDK 72   // padded LDS K-stride (144 B = 36 banks; 16B aligned)

template <bool STORE_BF16>
__global__ __launch_bounds__(256) void gemm_bf16_kernel(
    const u16* __restrict__ A, const u16* __restrict__ W,
    float* __restrict__ outF, u16* __restrict__ outB,
    int M, int N, int K) {
    __shared__ u16 Abuf[BM * LDK];
    __shared__ u16 Bbuf[BN * LDK];

    int tid  = threadIdx.x;
    int lane = tid & 31;
    int wave = tid >> 5;
    int mblk = blockIdx.x * BM;
    int nblk = blockIdx.y * BN;
    int wm = (wave >> 1) * 32;   // 0,32,64,96
    int wn = (wave & 1)  * 64;   // 0,64

    v8f zero = {};
    v8f acc[2][4];
#pragma unroll
    for (int i = 0; i < 2; ++i)
#pragma unroll
        for (int j = 0; j < 4; ++j) acc[i][j] = zero;

    for (int kb = 0; kb < K; kb += BK) {
        // Stage 128x64 A and B slabs into LDS: 1024 x 16B chunks each,
        // 4 chunks per thread per matrix.
#pragma unroll
        for (int c = 0; c < 4; ++c) {
            int idx = tid + c * 256;          // 0..1023
            int row = idx >> 3;               // 0..127
            int col = (idx & 7) << 3;         // 0,8,..,56
            copy16_g2l(A + (size_t)(mblk + row) * K + kb + col,
                       Abuf + row * LDK + col);
            copy16_g2l(W + (size_t)(nblk + row) * K + kb + col,
                       Bbuf + row * LDK + col);
        }
        // Prefetch next K-slab into L2 (CDNA5 global_prefetch_b8).
        if (kb + BK < K) {
            __builtin_prefetch(A + (size_t)(mblk + (tid >> 1)) * K + kb + BK, 0, 1);
            __builtin_prefetch(W + (size_t)(nblk + (tid >> 1)) * K + kb + BK, 0, 1);
        }
        async_wait_all();
        __syncthreads();

#pragma unroll
        for (int ks = 0; ks < BK; ks += 32) {
            v16bf a0 = load_fragA(Abuf, LDK, wm,      ks, lane);
            v16bf a1 = load_fragA(Abuf, LDK, wm + 16, ks, lane);
            v16bf b0 = load_fragB(Bbuf, LDK, wn,      ks, lane);
            v16bf b1 = load_fragB(Bbuf, LDK, wn + 16, ks, lane);
            v16bf b2 = load_fragB(Bbuf, LDK, wn + 32, ks, lane);
            v16bf b3 = load_fragB(Bbuf, LDK, wn + 48, ks, lane);
            acc[0][0] = wmma_bf16(a0, b0, acc[0][0]);
            acc[0][1] = wmma_bf16(a0, b1, acc[0][1]);
            acc[0][2] = wmma_bf16(a0, b2, acc[0][2]);
            acc[0][3] = wmma_bf16(a0, b3, acc[0][3]);
            acc[1][0] = wmma_bf16(a1, b0, acc[1][0]);
            acc[1][1] = wmma_bf16(a1, b1, acc[1][1]);
            acc[1][2] = wmma_bf16(a1, b2, acc[1][2]);
            acc[1][3] = wmma_bf16(a1, b3, acc[1][3]);
        }
        __syncthreads();
    }

    int n  = lane & 15;
    int mh = (lane >> 4) << 3;
#pragma unroll
    for (int i = 0; i < 2; ++i) {
#pragma unroll
        for (int r = 0; r < 8; ++r) {
            int mm = mblk + wm + i * 16 + mh + r;
#pragma unroll
            for (int j = 0; j < 4; ++j) {
                size_t o = (size_t)mm * N + nblk + wn + j * 16 + n;
                if constexpr (STORE_BF16) outB[o] = f2bf(acc[i][j][r]);
                else                      outF[o] = acc[i][j][r];
            }
        }
    }
}

// ---------------------------------------------------------------------------
// RoPE (interleaved, D=1024) + per-head assembly:
//   out[b,t,h,0:64]   = cpart[b,t,64h:64h+64]
//   out[b,t,h,64:128] = rope(rlin)[b,t,64h:64h+64]
// One thread per rotary pair. B*T = 4096 tokens, T = 2048.
// ---------------------------------------------------------------------------
__global__ void rope_assemble_kernel(const u16* __restrict__ cpart,
                                     const u16* __restrict__ rlin,
                                     u16* __restrict__ out, int npairs) {
    int i = blockIdx.x * blockDim.x + threadIdx.x;
    if (i >= npairs) return;
    int p  = i & 511;        // pair index within token (D/2 = 512)
    int bt = i >> 9;         // flat token index
    int t  = bt & 2047;      // position (T = 2048)

    // freq = theta^(-2p/D) = exp(-ln(1e4) * 2p / 1024)
    float freq = __expf(-9.210340371976184f * (float)(2 * p) * (1.0f / 1024.0f));
    float ang = (float)t * freq;
    float s, c;
    __sincosf(ang, &s, &c);

    float xe = bf2f(rlin[(size_t)bt * 1024 + 2 * p]);
    float xo = bf2f(rlin[(size_t)bt * 1024 + 2 * p + 1]);
    float re = xe * c - xo * s;
    float ro = xo * c + xe * s;

    int e0 = 2 * p;
    int hh = e0 >> 6;        // head (64 elems per head chunk)
    int d0 = e0 & 63;        // even, so e0,e0+1 share a head
    size_t ob = (size_t)bt * 2048 + (size_t)hh * 128;
    out[ob + 64 + d0]     = f2bf(re);
    out[ob + 64 + d0 + 1] = f2bf(ro);
    out[ob + d0]     = cpart[(size_t)bt * 1024 + e0];
    out[ob + d0 + 1] = cpart[(size_t)bt * 1024 + e0 + 1];
}

// ---------------------------------------------------------------------------
// V transpose: v[b,t,h*128+d] -> vt[((b*16+h)*128+d)*2048 + t]  (dim-major)
// ---------------------------------------------------------------------------
__global__ void transpose_v_kernel(const u16* __restrict__ v,
                                   u16* __restrict__ vt, int total) {
    int i = blockIdx.x * blockDim.x + threadIdx.x;
    if (i >= total) return;
    int t = i & 2047;
    int r = i >> 11;
    int d = r & 127; r >>= 7;
    int hh = r & 15;
    int b  = r >> 4;
    vt[i] = v[((size_t)b * 2048 + t) * 2048 + (size_t)hh * 128 + d];
}

// ---------------------------------------------------------------------------
// Attention: one block per (b, h, 16-query tile). 256 threads = 8 waves.
// Phase 1: causal score tiles (WMMA) -> dynamic LDS (16 x 2048 f32).
// Phase 2: masked softmax per row (16 lanes/row, shuffle reductions).
// Phase 3: PV WMMAs (A from LDS probs, B from dim-major V), store bf16.
// ---------------------------------------------------------------------------
__global__ __launch_bounds__(256) void flash_attn_kernel(
    const u16* __restrict__ q, const u16* __restrict__ k,
    const u16* __restrict__ vt, u16* __restrict__ attn) {
    extern __shared__ float Slds[];       // 16 rows x 2048 cols (128 KB)
    __shared__ float rowsum[16];

    int lane = threadIdx.x & 31;
    int wave = threadIdx.x >> 5;
    int qt = blockIdx.x & 127;            // T/16 = 128 query tiles
    int h  = (blockIdx.x >> 7) & 15;
    int b  = blockIdx.x >> 11;
    int q0 = qt * 16;
    const int T = 2048, Cdim = 2048, DHd = 128;

    const u16* qbase = q + ((size_t)b * T) * Cdim + (size_t)h * DHd;
    const u16* kbase = k + ((size_t)b * T) * Cdim + (size_t)h * DHd;

    // Q fragments for this 16-row tile (K = 128 head dims = 4 x 32)
    v16bf qa0 = load_fragA(qbase, Cdim, q0, 0,  lane);
    v16bf qa1 = load_fragA(qbase, Cdim, q0, 32, lane);
    v16bf qa2 = load_fragA(qbase, Cdim, q0, 64, lane);
    v16bf qa3 = load_fragA(qbase, Cdim, q0, 96, lane);

    const float scale = 0.08838834764831845f;   // 1/sqrt(128)
    int nkt = qt + 1;                           // causal: key tiles needed
    for (int kt = wave; kt < nkt; kt += 8) {
        v8f acc = {};
        acc = wmma_bf16(qa0, load_fragB(kbase, Cdim, kt * 16, 0,  lane), acc);
        acc = wmma_bf16(qa1, load_fragB(kbase, Cdim, kt * 16, 32, lane), acc);
        acc = wmma_bf16(qa2, load_fragB(kbase, Cdim, kt * 16, 64, lane), acc);
        acc = wmma_bf16(qa3, load_fragB(kbase, Cdim, kt * 16, 96, lane), acc);
        int n = lane & 15, mh = (lane >> 4) << 3;
#pragma unroll
        for (int r = 0; r < 8; ++r)
            Slds[(mh + r) * 2048 + kt * 16 + n] = acc[r] * scale;
    }
    __syncthreads();

    // Softmax: row = 2*wave + (lane>=16), 16 lanes per row
    int row  = wave * 2 + (lane >> 4);
    int j    = lane & 15;
    int qpos = q0 + row;
    int nk   = q0 + 16;                       // keys written to LDS
    int nk32 = (nk + 31) & ~31;               // padded to 32 for PV k-steps
    float* S = Slds + row * 2048;
    for (int kk = nk + j; kk < nk32; kk += 16) S[kk] = 0.0f;  // zero pad
    float m = -3.4e38f;
    for (int kk = j; kk <= qpos; kk += 16) m = fmaxf(m, S[kk]);
#pragma unroll
    for (int off = 1; off < 16; off <<= 1) m = fmaxf(m, __shfl_xor(m, off, 32));
    float sum = 0.0f;
    for (int kk = j; kk < nk; kk += 16) {
        float pv = (kk <= qpos) ? __expf(S[kk] - m) : 0.0f;
        S[kk] = pv;                           // unnormalized prob
        sum += pv;
    }
#pragma unroll
    for (int off = 1; off < 16; off <<= 1) sum += __shfl_xor(sum, off, 32);
    if (j == 0) rowsum[row] = sum;
    __syncthreads();

    // PV: wave handles one 16-dim slice of the 128-dim head
    int nd = wave * 16;
    const u16* vbase = vt + ((size_t)((b * 16 + h) * 128)) * T;  // row = dim
    v8f acc = {};
    int mrow = lane & 15;
    int hs   = (lane >> 4) << 3;
    for (int kc = 0; kc < nk32; kc += 32) {
        const float* Sp = Slds + mrow * 2048 + kc;
        Frag fa;
#pragma unroll
        for (int e = 0; e < 8; ++e) fa.u[e] = f2bf(Sp[hs + e]);
#pragma unroll
        for (int e = 0; e < 8; ++e) fa.u[8 + e] = f2bf(Sp[16 + hs + e]);
        v16bf bv = load_fragB(vbase, T, nd, kc, lane);
        acc = wmma_bf16(fa.v, bv, acc);
    }
    int n = lane & 15, mh = (lane >> 4) << 3;
#pragma unroll
    for (int r = 0; r < 8; ++r) {
        int mm = mh + r;
        float o = acc[r] / rowsum[mm];
        int t = q0 + mm;
        attn[((size_t)b * T + t) * Cdim + (size_t)h * DHd + nd + n] = f2bf(o);
    }
}

// ---------------------------------------------------------------------------
// Host orchestration
// ---------------------------------------------------------------------------
extern "C" void kernel_launch(void* const* d_in, const int* in_sizes, int n_in,
                              void* d_out, int out_size, void* d_ws, size_t ws_size,
                              hipStream_t stream) {
    const float* x     = (const float*)d_in[0];
    const float* W_kvD = (const float*)d_in[1];
    const float* W_qD  = (const float*)d_in[2];
    const float* W_kU  = (const float*)d_in[3];
    const float* W_vU  = (const float*)d_in[4];
    const float* W_qU  = (const float*)d_in[5];
    const float* W_rk  = (const float*)d_in[6];
    const float* W_rq  = (const float*)d_in[7];
    const float* W_o   = (const float*)d_in[8];
    float* out = (float*)d_out;

    const int B = 2, T = 2048, C = 2048, L = 512, Hh = 16;
    const int BT = B * T;  // 4096

    char* p = (char*)d_ws;
    auto carve = [&](size_t elems) {
        u16* r = (u16*)p;
        p += (elems * sizeof(u16) + 255) & ~(size_t)255;
        return r;
    };
    u16* x_bf    = carve((size_t)BT * C);
    u16* WkvD_bf = carve((size_t)L * C);
    u16* WqD_bf  = carve((size_t)L * C);
    u16* WkU_bf  = carve((size_t)1024 * L);
    u16* WvU_bf  = carve((size_t)C * L);
    u16* WqU_bf  = carve((size_t)1024 * L);
    u16* Wrk_bf  = carve((size_t)1024 * C);
    u16* Wrq_bf  = carve((size_t)1024 * L);
    u16* Wo_bf   = carve((size_t)C * C);
    u16* kvd_bf  = carve((size_t)BT * L);
    u16* qd_bf   = carve((size_t)BT * L);
    u16* kc_bf   = carve((size_t)BT * 1024);
    u16* qc_bf   = carve((size_t)BT * 1024);
    u16* kr_bf   = carve((size_t)BT * 1024);
    u16* qr_bf   = carve((size_t)BT * 1024);
    u16* v_bf    = carve((size_t)BT * C);
    u16* vt_bf   = carve((size_t)BT * C);
    u16* qq_bf   = carve((size_t)BT * C);
    u16* kk_bf   = carve((size_t)BT * C);
    u16* attn_bf = carve((size_t)BT * C);

    auto cvt = [&](const float* s, u16* d, int n) {
        cvt_f32_to_bf16_kernel<<<(n + 255) / 256, 256, 0, stream>>>(s, d, n);
    };
    cvt(x, x_bf, BT * C);
    cvt(W_kvD, WkvD_bf, L * C);
    cvt(W_qD,  WqD_bf,  L * C);
    cvt(W_kU,  WkU_bf,  1024 * L);
    cvt(W_vU,  WvU_bf,  C * L);
    cvt(W_qU,  WqU_bf,  1024 * L);
    cvt(W_rk,  Wrk_bf,  1024 * C);
    cvt(W_rq,  Wrq_bf,  1024 * L);
    cvt(W_o,   Wo_bf,   C * C);

    auto gemmB = [&](const u16* A, const u16* W, u16* D, int M, int N, int K) {
        dim3 g(M / 128, N / 128);
        gemm_bf16_kernel<true><<<g, 256, 0, stream>>>(A, W, nullptr, D, M, N, K);
    };
    gemmB(x_bf,   WkvD_bf, kvd_bf, BT, L,    C);   // kv_d
    gemmB(x_bf,   WqD_bf,  qd_bf,  BT, L,    C);   // q_d
    gemmB(x_bf,   Wrk_bf,  kr_bf,  BT, 1024, C);   // k_r (pre-rope)
    gemmB(qd_bf,  Wrq_bf,  qr_bf,  BT, 1024, L);   // q_r (pre-rope)
    gemmB(kvd_bf, WkU_bf,  kc_bf,  BT, 1024, L);   // k_c
    gemmB(qd_bf,  WqU_bf,  qc_bf,  BT, 1024, L);   // q_c
    gemmB(kvd_bf, WvU_bf,  v_bf,   BT, C,    L);   // v

    int npairs = BT * 512;
    rope_assemble_kernel<<<(npairs + 255) / 256, 256, 0, stream>>>(kc_bf, kr_bf, kk_bf, npairs);
    rope_assemble_kernel<<<(npairs + 255) / 256, 256, 0, stream>>>(qc_bf, qr_bf, qq_bf, npairs);

    int tot = BT * C;
    transpose_v_kernel<<<(tot + 255) / 256, 256, 0, stream>>>(v_bf, vt_bf, tot);

    flash_attn_kernel<<<B * Hh * (T / 16), 256, 16 * 2048 * sizeof(float), stream>>>(
        qq_bf, kk_bf, vt_bf, attn_bf);

    dim3 g(BT / 128, C / 128);
    gemm_bf16_kernel<false><<<g, 256, 0, stream>>>(attn_bf, Wo_bf, out, nullptr, BT, C, C);
}
